// GCCN_3_63917703299195
// MI455X (gfx1250) — compile-verified, matrix operand order checked by reading, not run on hardware
//
#include <hip/hip_runtime.h>

#define NBINS 80

typedef __attribute__((ext_vector_type(16))) _Float16 v16h;
typedef __attribute__((ext_vector_type(8)))  _Float16 v8h;
typedef __attribute__((ext_vector_type(2)))  _Float16 v2h;
typedef __attribute__((ext_vector_type(8)))  float    v8f;
typedef __attribute__((ext_vector_type(4)))  float    v4f;

// ---------------------------------------------------------------------------
// CDNA5 async global->LDS copy (ASYNCcnt-tracked, no VGPR round trip).
// ---------------------------------------------------------------------------
__device__ __forceinline__ void async_copy_b128(unsigned lds_addr,
                                                const void* gbase,
                                                unsigned voff) {
  asm volatile("global_load_async_to_lds_b128 %0, %1, %2"
               :: "v"(lds_addr), "v"(voff), "s"(gbase) : "memory");
}
__device__ __forceinline__ void wait_async0() {
  asm volatile("s_wait_asynccnt 0x0" ::: "memory");
}
__device__ __forceinline__ unsigned lds_addr_of(const void* p) {
  return (unsigned)(unsigned long long)p;   // low 32 bits of LDS aperture addr
}

// A fragment (16x32 f16 x2 K-steps) via uniform base + 32-bit byte offset.
// lane<16 -> K {0..7,16..23}; lane>=16 -> K {8..15,24..31}
__device__ __forceinline__ void load_a_off(v16h& s0, v16h& s1,
                                           const _Float16* base, unsigned off) {
  const char* p = (const char*)base + off;
  *(v8h*)&s0       = *(const v8h*)(p);
  *((v8h*)&s0 + 1) = *(const v8h*)(p + 32);
  *(v8h*)&s1       = *(const v8h*)(p + 64);
  *((v8h*)&s1 + 1) = *(const v8h*)(p + 96);
}

__device__ __forceinline__ v16h load_b(const _Float16* base) {
  v16h r;
  *(v8h*)&r       = *(const v8h*)base;
  *((v8h*)&r + 1) = *(const v8h*)(base + 16);
  return r;
}

#define WMMA_F16(A, B, C) \
  __builtin_amdgcn_wmma_f32_16x16x32_f16(false, (A), false, (B), (short)0, (C), false, false)

// One bin, two M-tiles: each B fragment loaded once, used by 2 WMMAs.
// c[0..NT-1] = tile0 accumulators, c[NT..2NT-1] = tile1 accumulators.
template <int NT>
__device__ __forceinline__ void bin_mma2(const _Float16* wb, int m, int hi,
                                         const v16h& x00, const v16h& x01,
                                         const v16h& x10, const v16h& x11,
                                         v8f* c) {
#pragma unroll
  for (int nt = 0; nt < NT; ++nt) {
    v16h bf = load_b(wb + (nt * 16 + m) * 64 + 8 * hi);
    c[nt]      = WMMA_F16(x00, bf, c[nt]);
    c[NT + nt] = WMMA_F16(x10, bf, c[NT + nt]);
  }
#pragma unroll
  for (int nt = 0; nt < NT; ++nt) {
    v16h bf = load_b(wb + (nt * 16 + m) * 64 + 32 + 8 * hi);
    c[nt]      = WMMA_F16(x01, bf, c[nt]);
    c[NT + nt] = WMMA_F16(x11, bf, c[NT + nt]);
  }
}

// ---------------------------------------------------------------------------
// Stage 1: h = relu(x @ W1) -> f16 [N,64].  Tiny (61 MFLOP) -> VALU.
// ---------------------------------------------------------------------------
__global__ void prep_h(const float* __restrict__ x, const float* __restrict__ W1,
                       _Float16* __restrict__ h1, int total) {
  int idx = blockIdx.x * blockDim.x + threadIdx.x;
  if (idx >= total) return;
  int n = idx >> 6, o = idx & 63;
  const float* xr = x + (size_t)n * 16;
  float acc = 0.f;
#pragma unroll
  for (int i = 0; i < 16; ++i) acc = fmaf(xr[i], W1[i * 64 + o], acc);
  h1[idx] = (_Float16)fmaxf(acc, 0.f);
}

// ---------------------------------------------------------------------------
// Weight prep: Wg[b][i][o] -> wT[b][o][i] in f16 (per-lane-contiguous B frags)
// ---------------------------------------------------------------------------
__global__ void prep_w(const float* __restrict__ Wg1, const float* __restrict__ Wg2,
                       _Float16* __restrict__ w1T, _Float16* __restrict__ w2T) {
  int idx = blockIdx.x * blockDim.x + threadIdx.x;
  const int n1 = NBINS * 64 * 64;
  const int n2 = NBINS * 32 * 64;
  if (idx < n1) {
    int b = idx / (64 * 64), r = idx % (64 * 64);
    int o = r >> 6, i = r & 63;
    w1T[idx] = (_Float16)Wg1[(size_t)b * 64 * 64 + i * 64 + o];
  } else if (idx < n1 + n2) {
    int k = idx - n1;
    int b = k / (32 * 64), r = k % (32 * 64);
    int o = r >> 6, i = r & 63;
    w2T[k] = (_Float16)Wg2[(size_t)b * 64 * 32 + i * 32 + o];
  }
}

// ---------------------------------------------------------------------------
// GC1: 4 waves x 32 vertices per block; 2x-unrolled ping-pong bin loop;
// double-buffered async LDS weight staging; B fragments reused across M-tiles.
// ---------------------------------------------------------------------------
__global__ __launch_bounds__(128) void gc1(
    const _Float16* __restrict__ h1, const _Float16* __restrict__ w1T,
    const float* __restrict__ b1, const int* __restrict__ conn,
    _Float16* __restrict__ h2, int Nv) {
  __shared__ _Float16 lds_w[2][64 * 64];   // 2 x 8 KB bin weights
  __shared__ float    lds_c[4][32 * 64];   // per-wave C staging (32 KB)

  const int tid  = threadIdx.x;
  const int wave = tid >> 5;
  const int lane = tid & 31;
  const int m    = lane & 15;
  const int hi   = lane >> 4;
  const int v0   = (blockIdx.x * 4 + wave) * 32;
  int va = v0 + m;      if (va >= Nv) va = Nv - 1;   // tile0 row (clamped)
  int vb = v0 + 16 + m; if (vb >= Nv) vb = Nv - 1;   // tile1 row (clamped)
  const int* connRow0 = conn + (size_t)va * NBINS;
  const int* connRow1 = conn + (size_t)vb * NBINS;

  const unsigned lb0 = lds_addr_of(&lds_w[0][0]);
  const unsigned lb1 = lds_addr_of(&lds_w[1][0]);
  const unsigned t16 = (unsigned)(tid * 16);

  // ---- prologue: stage bin 0 (8 KB, 128 thr x 4 x b128), prefetch gather ----
#pragma unroll
  for (int j = 0; j < 4; ++j)
    async_copy_b128(lb0 + t16 + j * 2048, (const char*)w1T, t16 + j * 2048);
  int srowA0 = connRow0[0], srowA1 = connRow1[0];
  int srowB0 = connRow0[1], srowB1 = connRow1[1];
  v16h aA00, aA01, aA10, aA11, aB00, aB01, aB10, aB11;
  load_a_off(aA00, aA01, h1, (unsigned)srowA0 * 128u + 16u * hi);
  load_a_off(aA10, aA11, h1, (unsigned)srowA1 * 128u + 16u * hi);

  v8f c[8] = {};

  for (int b = 0; b < NBINS; b += 2) {
    // ===== even phase: bin b from buf0 / aA =====
    wait_async0();
    __syncthreads();
    {   // stage bin b+1 -> buf1
      const char* src = (const char*)(w1T + (size_t)(b + 1) * 64 * 64);
#pragma unroll
      for (int j = 0; j < 4; ++j)
        async_copy_b128(lb1 + t16 + j * 2048, src, t16 + j * 2048);
    }
    load_a_off(aB00, aB01, h1, (unsigned)srowB0 * 128u + 16u * hi);
    load_a_off(aB10, aB11, h1, (unsigned)srowB1 * 128u + 16u * hi);
    {
      int nb = (b + 2 < NBINS) ? (b + 2) : (NBINS - 1);
      srowA0 = connRow0[nb]; srowA1 = connRow1[nb];
    }
    bin_mma2<4>(lds_w[0], m, hi, aA00, aA01, aA10, aA11, c);

    // ===== odd phase: bin b+1 from buf1 / aB =====
    wait_async0();
    __syncthreads();
    if (b + 2 < NBINS) {
      const char* src = (const char*)(w1T + (size_t)(b + 2) * 64 * 64);
#pragma unroll
      for (int j = 0; j < 4; ++j)
        async_copy_b128(lb0 + t16 + j * 2048, src, t16 + j * 2048);
      load_a_off(aA00, aA01, h1, (unsigned)srowA0 * 128u + 16u * hi);
      load_a_off(aA10, aA11, h1, (unsigned)srowA1 * 128u + 16u * hi);
      int nb = (b + 3 < NBINS) ? (b + 3) : (NBINS - 1);
      srowB0 = connRow0[nb]; srowB1 = connRow1[nb];
    }
    bin_mma2<4>(lds_w[1], m, hi, aB00, aB01, aB10, aB11, c);
  }

  // ---- epilogue: stage C in LDS, coalesced bias+relu+f16 store ----
  float* cs = lds_c[wave];
#pragma unroll
  for (int r = 0; r < 8; ++r) {
    int rm = r + 8 * hi;
#pragma unroll
    for (int nt = 0; nt < 4; ++nt) {
      cs[rm * 64        + nt * 16 + m] = c[nt][r];        // tile0: rows 0..15
      cs[(16 + rm) * 64 + nt * 16 + m] = c[4 + nt][r];    // tile1: rows 16..31
    }
  }
  __syncthreads();
  const float bb0 = b1[lane * 2];
  const float bb1 = b1[lane * 2 + 1];
#pragma unroll
  for (int row = 0; row < 32; ++row) {
    float t0 = fmaxf(cs[row * 64 + lane * 2]     + bb0, 0.f);
    float t1 = fmaxf(cs[row * 64 + lane * 2 + 1] + bb1, 0.f);
    v2h pk = { (_Float16)t0, (_Float16)t1 };
    if (v0 + row < Nv)
      *(v2h*)(h2 + (size_t)(v0 + row) * 64 + lane * 2) = pk;
  }
}

// ---------------------------------------------------------------------------
// GC2: same pipeline, N=32; epilogue adds bias + L2-normalizes (1 row/lane).
// ---------------------------------------------------------------------------
__global__ __launch_bounds__(128) void gc2(
    const _Float16* __restrict__ h2, const _Float16* __restrict__ w2T,
    const float* __restrict__ b2, const int* __restrict__ conn,
    float* __restrict__ out, int Nv) {
  __shared__ _Float16 lds_w[2][32 * 64];   // 2 x 4 KB
  __shared__ float    lds_c[4][32 * 32];   // 16 KB

  const int tid  = threadIdx.x;
  const int wave = tid >> 5;
  const int lane = tid & 31;
  const int m    = lane & 15;
  const int hi   = lane >> 4;
  const int v0   = (blockIdx.x * 4 + wave) * 32;
  int va = v0 + m;      if (va >= Nv) va = Nv - 1;
  int vb = v0 + 16 + m; if (vb >= Nv) vb = Nv - 1;
  const int* connRow0 = conn + (size_t)va * NBINS;
  const int* connRow1 = conn + (size_t)vb * NBINS;

  const unsigned lb0 = lds_addr_of(&lds_w[0][0]);
  const unsigned lb1 = lds_addr_of(&lds_w[1][0]);
  const unsigned t16 = (unsigned)(tid * 16);

#pragma unroll
  for (int j = 0; j < 2; ++j)
    async_copy_b128(lb0 + t16 + j * 2048, (const char*)w2T, t16 + j * 2048);
  int srowA0 = connRow0[0], srowA1 = connRow1[0];
  int srowB0 = connRow0[1], srowB1 = connRow1[1];
  v16h aA00, aA01, aA10, aA11, aB00, aB01, aB10, aB11;
  load_a_off(aA00, aA01, h2, (unsigned)srowA0 * 128u + 16u * hi);
  load_a_off(aA10, aA11, h2, (unsigned)srowA1 * 128u + 16u * hi);

  v8f c[4] = {};

  for (int b = 0; b < NBINS; b += 2) {
    // ===== even phase =====
    wait_async0();
    __syncthreads();
    {
      const char* src = (const char*)(w2T + (size_t)(b + 1) * 32 * 64);
#pragma unroll
      for (int j = 0; j < 2; ++j)
        async_copy_b128(lb1 + t16 + j * 2048, src, t16 + j * 2048);
    }
    load_a_off(aB00, aB01, h2, (unsigned)srowB0 * 128u + 16u * hi);
    load_a_off(aB10, aB11, h2, (unsigned)srowB1 * 128u + 16u * hi);
    {
      int nb = (b + 2 < NBINS) ? (b + 2) : (NBINS - 1);
      srowA0 = connRow0[nb]; srowA1 = connRow1[nb];
    }
    bin_mma2<2>(lds_w[0], m, hi, aA00, aA01, aA10, aA11, c);

    // ===== odd phase =====
    wait_async0();
    __syncthreads();
    if (b + 2 < NBINS) {
      const char* src = (const char*)(w2T + (size_t)(b + 2) * 32 * 64);
#pragma unroll
      for (int j = 0; j < 2; ++j)
        async_copy_b128(lb0 + t16 + j * 2048, src, t16 + j * 2048);
      load_a_off(aA00, aA01, h2, (unsigned)srowA0 * 128u + 16u * hi);
      load_a_off(aA10, aA11, h2, (unsigned)srowA1 * 128u + 16u * hi);
      int nb = (b + 3 < NBINS) ? (b + 3) : (NBINS - 1);
      srowB0 = connRow0[nb]; srowB1 = connRow1[nb];
    }
    bin_mma2<2>(lds_w[1], m, hi, aB00, aB01, aB10, aB11, c);
  }

  float* cs = lds_c[wave];
#pragma unroll
  for (int r = 0; r < 8; ++r) {
    int rm = r + 8 * hi;
#pragma unroll
    for (int nt = 0; nt < 2; ++nt) {
      cs[rm * 32        + nt * 16 + m] = c[nt][r];      // tile0: rows 0..15
      cs[(16 + rm) * 32 + nt * 16 + m] = c[2 + nt][r];  // tile1: rows 16..31
    }
  }
  __syncthreads();

  {   // one output row per lane: bias, L2-norm, store 32 f32
    const int row = lane;
    const int vv  = v0 + row;
    float vals[32];
    float ss = 0.f;
#pragma unroll
    for (int k = 0; k < 32; ++k) {
      float t = cs[row * 32 + k] + b2[k];
      vals[k] = t;
      ss = fmaf(t, t, ss);
    }
    const float inv = 1.0f / sqrtf(ss);
    if (vv < Nv) {
      v4f* op = (v4f*)(out + (size_t)vv * 32);
#pragma unroll
      for (int q = 0; q < 8; ++q) {
        v4f t = { vals[4 * q + 0] * inv, vals[4 * q + 1] * inv,
                  vals[4 * q + 2] * inv, vals[4 * q + 3] * inv };
        op[q] = t;
      }
    }
  }
}

// ---------------------------------------------------------------------------
extern "C" void kernel_launch(void* const* d_in, const int* in_sizes, int n_in,
                              void* d_out, int out_size, void* d_ws, size_t ws_size,
                              hipStream_t stream) {
  (void)n_in; (void)out_size; (void)ws_size;
  const float* x    = (const float*)d_in[0];
  const float* W1   = (const float*)d_in[1];
  const float* Wg1  = (const float*)d_in[2];
  const float* b1   = (const float*)d_in[3];
  const float* Wg2  = (const float*)d_in[4];
  const float* b2   = (const float*)d_in[5];
  const int*   conn = (const int*)d_in[6];
  float* out = (float*)d_out;
  const int Nv = in_sizes[0] / 16;

  _Float16* h1  = (_Float16*)d_ws;                 // Nv*64 f16
  _Float16* h2  = h1  + (size_t)Nv * 64;           // Nv*64 f16
  _Float16* w1T = h2  + (size_t)Nv * 64;           // 80*64*64 f16
  _Float16* w2T = w1T + (size_t)NBINS * 64 * 64;   // 80*32*64 f16

  {
    int total = Nv * 64;
    prep_h<<<(total + 255) / 256, 256, 0, stream>>>(x, W1, h1, total);
  }
  {
    int total = NBINS * 64 * 64 + NBINS * 32 * 64;
    prep_w<<<(total + 255) / 256, 256, 0, stream>>>(Wg1, Wg2, w1T, w2T);
  }
  const int blocks = (Nv + 127) / 128;   // 4 waves x 32 vertices per block
  gc1<<<blocks, 128, 0, stream>>>(h1, w1T, b1, conn, h2, Nv);
  gc2<<<blocks, 128, 0, stream>>>(h2, w2T, b2, conn, out, Nv);
}
